// HierarchicalAttention_15805479649728
// MI455X (gfx1250) — matrix operand order repaired and checked by reading, not compiled, not run on hardware
//
#include <hip/hip_runtime.h>
#include <math.h>

// ---------------------------------------------------------------------------
// HierarchicalAttention forward for MI455X (gfx1250), bf16 WMMA pipeline.
// B=2 S=4096 D=1024 H=16 HD=64 WIN=1024 G=256, scale = 1/8.
// ---------------------------------------------------------------------------

typedef __attribute__((ext_vector_type(16))) __bf16 bf16x16;
typedef __attribute__((ext_vector_type(8)))  float  f32x8;

#define CB 2
#define CS 4096
#define CD 1024
#define CH 16
#define CHD 64
#define CG 256
#define CWIN 1024
#define CNW 4

// ---- helpers ---------------------------------------------------------------

__device__ __forceinline__ unsigned short f2bf(float f) {
    unsigned u = __float_as_uint(f);
    u += 0x7FFFu + ((u >> 16) & 1u);   // round-to-nearest-even
    return (unsigned short)(u >> 16);
}

__device__ __forceinline__ f32x8 wmma_bf16(bf16x16 a, bf16x16 b, f32x8 c) {
    return __builtin_amdgcn_wmma_f32_16x16x32_bf16(
        /*neg_a=*/false, a, /*neg_b=*/false, b,
        /*c_mod=*/(short)0, c, /*reuse_a=*/false, /*reuse_b=*/false);
}

// A fragment (16x32 bf16, row-major source). p = &A[row_of_lane][k0].
// ISA layout: lanes 0-15 hold K = k0 + {0..7, 16..23}; lanes 16-31 K = k0 + {8..15, 24..31}.
__device__ __forceinline__ bf16x16 ld_a(const unsigned short* p, int half) {
    union { uint4 u[2]; bf16x16 v; } t;
    t.u[0] = *(const uint4*)(p + half * 8);
    t.u[1] = *(const uint4*)(p + 16 + half * 8);
    return t.v;
}

// B fragment (32x16 bf16). Source stored as Bt[N][Kd] (row = output column n).
// p = &Bt[n_of_lane][k0]. ISA layout: lanes 0-15 hold K = k0 + {0..7, 16..23}; lanes 16-31 K = k0 + {8..15, 24..31}.
__device__ __forceinline__ bf16x16 ld_b(const unsigned short* p, int half) {
    union { uint4 u[2]; bf16x16 v; } t;
    t.u[0] = *(const uint4*)(p + half * 8);
    t.u[1] = *(const uint4*)(p + 16 + half * 8);
    return t.v;
}

__device__ __forceinline__ unsigned ordf(float f) {
    unsigned u = __float_as_uint(f);
    return (u & 0x80000000u) ? ~u : (u | 0x80000000u);
}

// ---- elementwise converts --------------------------------------------------

__global__ void cvt_bf16_kernel(const float* __restrict__ src,
                                unsigned short* __restrict__ dst, int n) {
    int i = blockIdx.x * blockDim.x + threadIdx.x;
    if (i < n) dst[i] = f2bf(src[i]);
}

// wt[n*K + k] = w[k*N + n]   (store W^T in bf16 for contiguous B-fragments)
__global__ void cvt_t_kernel(const float* __restrict__ w,
                             unsigned short* __restrict__ wt, int K, int N) {
    int i = blockIdx.x * blockDim.x + threadIdx.x;
    if (i >= K * N) return;
    int n = i % N, k = i / N;
    wt[(size_t)n * K + k] = f2bf(w[i]);
}

// ---- token scorer: gelu(x@ws1+bs1)@ws2+bs2, triu([B,S],1) mask -------------

__global__ __launch_bounds__(256) void score_kernel(
    const float* __restrict__ x, const float* __restrict__ ws1,
    const float* __restrict__ bs1, const float* __restrict__ ws2,
    const float* __restrict__ bs2, float* __restrict__ scores) {
    __shared__ float xs[CD];
    __shared__ float red[256];
    int bs = blockIdx.x;                 // b*S + s
    int b = bs >> 12, s = bs & (CS - 1);
    const float* xr = x + (size_t)bs * CD;
    for (int k = threadIdx.x; k < CD; k += 256) xs[k] = xr[k];
    __syncthreads();
    int j = threadIdx.x;                 // hidden unit (256 of them)
    float acc = bs1[j];
    for (int k = 0; k < CD; ++k) acc = fmaf(xs[k], ws1[k * 256 + j], acc);
    float g = 0.5f * acc * (1.0f + erff(acc * 0.7071067811865476f)); // exact gelu
    red[j] = g * ws2[j];
    __syncthreads();
    for (int off = 128; off > 0; off >>= 1) {
        if (threadIdx.x < off) red[threadIdx.x] += red[threadIdx.x + off];
        __syncthreads();
    }
    if (threadIdx.x == 0) {
        float sc = red[0] + bs2[0];
        if (s > b) sc = -3.4028234663852886e38f;  // finfo(f32).min quirk mask
        scores[bs] = sc;
    }
}

// ---- top-k by rank (descending score, ties -> lower index first) -----------

__global__ void topk_kernel(const float* __restrict__ scores, int* __restrict__ idx) {
    int t = blockIdx.x * blockDim.x + threadIdx.x;
    if (t >= CB * CS) return;
    int b = t >> 12, s = t & (CS - 1);
    const float* sc = scores + (size_t)b * CS;
    unsigned long long mykey =
        ((unsigned long long)ordf(sc[s]) << 32) | (unsigned)(CS - 1 - s);
    int rank = 0;
    for (int j = 0; j < CS; ++j) {
        unsigned long long kk =
            ((unsigned long long)ordf(sc[j]) << 32) | (unsigned)(CS - 1 - j);
        rank += (kk > mykey) ? 1 : 0;
    }
    if (rank < CG) idx[b * CG + rank] = s;
}

__global__ void gather_kernel(const unsigned short* __restrict__ xbf,
                              const int* __restrict__ idx,
                              unsigned short* __restrict__ gtok) {
    int i = blockIdx.x * blockDim.x + threadIdx.x;   // over B*G*D
    if (i >= CB * CG * CD) return;
    int c = i & (CD - 1);
    int g = (i >> 10) & (CG - 1);
    int b = i >> 18;
    int s = idx[b * CG + g];
    gtok[i] = xbf[((size_t)b * CS + s) * CD + c];
}

// ---- generic bf16 WMMA GEMM: C[M,N] = A[M,Kd] * Bt[N,Kd]^T -----------------
// mode 0: bf16 out, head layout   [((b*H+h)*S_ + s)*64 + hd], row = b*S_+s
// mode 1: bf16 out, head layout T [((b*H+h)*64 + hd)*S_ + s]
// mode 2: f32  out, row-major     [row*N + col]

__global__ __launch_bounds__(256) void gemm_bf16_kernel(
    const unsigned short* __restrict__ A, const unsigned short* __restrict__ Bt,
    int M, int N, int Kd, int mode, float scale, int S_,
    unsigned short* __restrict__ outb, float* __restrict__ outf) {
    int lane = threadIdx.x & 31, w = threadIdx.x >> 5;
    int half = lane >> 4, ln = lane & 15;
    int mt = M >> 4;
    long tile = (long)blockIdx.x * 8 + w;
    int mtile = (int)(tile % mt);
    int ntile = (int)(tile / mt);
    int row0 = mtile * 16, col0 = ntile * 64;

    const unsigned short* arow = A + (size_t)(row0 + ln) * Kd;
    f32x8 c0 = {}, c1 = {}, c2 = {}, c3 = {};
    for (int k0 = 0; k0 < Kd; k0 += 32) {
        bf16x16 a = ld_a(arow + k0, half);
        const unsigned short* bb = Bt + (size_t)(col0 + ln) * Kd + k0;
        c0 = wmma_bf16(a, ld_b(bb, half), c0);
        c1 = wmma_bf16(a, ld_b(bb + (size_t)16 * Kd, half), c1);
        c2 = wmma_bf16(a, ld_b(bb + (size_t)32 * Kd, half), c2);
        c3 = wmma_bf16(a, ld_b(bb + (size_t)48 * Kd, half), c3);
    }
#pragma unroll
    for (int r = 0; r < 8; ++r) {
        int row = row0 + half * 8 + r;
        float vals[4] = { c0[r], c1[r], c2[r], c3[r] };
#pragma unroll
        for (int f = 0; f < 4; ++f) {
            int col = col0 + f * 16 + ln;
            float v = vals[f] * scale;
            if (mode == 2) {
                outf[(size_t)row * N + col] = v;
            } else {
                int b = row / S_, s = row % S_;
                int hh = col >> 6, hd = col & 63;
                size_t o = (mode == 0)
                    ? ((((size_t)b * CH + hh) * S_ + s) * 64 + hd)
                    : ((((size_t)b * CH + hh) * 64 + hd) * S_ + s);
                outb[o] = f2bf(v);
            }
        }
    }
}

// ---- flash-softmax block update (16 queries x 32 keys) ---------------------

__device__ __forceinline__ void flash_block(
    f32x8 s0, f32x8 s1, bool causal, int kb, int q0w, int half, int ln,
    const unsigned short* __restrict__ vrow0, size_t vld, int koff,
    float* mrow, float* lrow, f32x8& o0, f32x8& o1, f32x8& o2, f32x8& o3,
    unsigned short* lpw) {
    float p0[8], p1[8];
#pragma unroll
    for (int r = 0; r < 8; ++r) {
        float a = s0[r], c = s1[r];
        if (causal) {
            int qrow = q0w + half * 8 + r;
            if (kb + ln > qrow)      a = -__builtin_inff();
            if (kb + 16 + ln > qrow) c = -__builtin_inff();
        }
        float mx = fmaxf(a, c);
        mx = fmaxf(mx, __shfl_xor(mx, 1));
        mx = fmaxf(mx, __shfl_xor(mx, 2));
        mx = fmaxf(mx, __shfl_xor(mx, 4));
        mx = fmaxf(mx, __shfl_xor(mx, 8));
        float mn = fmaxf(mrow[r], mx);
        float scl = __expf(mrow[r] - mn);
        a = __expf(a - mn);
        c = __expf(c - mn);
        float sm = a + c;
        sm += __shfl_xor(sm, 1);
        sm += __shfl_xor(sm, 2);
        sm += __shfl_xor(sm, 4);
        sm += __shfl_xor(sm, 8);
        lrow[r] = lrow[r] * scl + sm;
        mrow[r] = mn;
        o0[r] *= scl; o1[r] *= scl; o2[r] *= scl; o3[r] *= scl;
        p0[r] = a; p1[r] = c;
    }
    // D-fragment (row = 8*half+r, col = ln) -> LDS -> A-fragment layout
#pragma unroll
    for (int r = 0; r < 8; ++r) {
        lpw[(half * 8 + r) * 32 + ln]      = f2bf(p0[r]);
        lpw[(half * 8 + r) * 32 + 16 + ln] = f2bf(p1[r]);
    }
    asm volatile("s_wait_dscnt 0" ::: "memory");
    bf16x16 pa = ld_a(lpw + ln * 32, half);            // 16x32, Kd=32
    o0 = wmma_bf16(pa, ld_b(vrow0 + (size_t)(ln)      * vld + koff, half), o0);
    o1 = wmma_bf16(pa, ld_b(vrow0 + (size_t)(16 + ln) * vld + koff, half), o1);
    o2 = wmma_bf16(pa, ld_b(vrow0 + (size_t)(32 + ln) * vld + koff, half), o2);
    o3 = wmma_bf16(pa, ld_b(vrow0 + (size_t)(48 + ln) * vld + koff, half), o3);
}

// ---- attention: local causal window + global tokens ------------------------

__global__ __launch_bounds__(256) void attn_kernel(
    const unsigned short* __restrict__ qm,   // [b,h,s,hd] bf16 (pre-scaled)
    const unsigned short* __restrict__ km,   // [b,h,s,hd] bf16
    const unsigned short* __restrict__ vt,   // [b,h,hd,s] bf16
    const unsigned short* __restrict__ gk,   // [b,h,g,hd] bf16
    const unsigned short* __restrict__ gvt,  // [b,h,hd,g] bf16
    unsigned short* __restrict__ attn) {     // [b,s,h*64+hd] bf16
    __shared__ __align__(16) unsigned short lp[8][512];
    int lane = threadIdx.x & 31, w = threadIdx.x >> 5;
    int half = lane >> 4, ln = lane & 15;
    int gtile = blockIdx.x * 8 + w;          // 8192 tiles total
    int tq  = gtile & 63;                    // 64 query tiles / window
    int widx = gtile >> 6;                   // (b,h,win)
    int win = widx & (CNW - 1);
    int h   = (widx >> 2) & (CH - 1);
    int b   = widx >> 6;
    int q0w = tq * 16;
    size_t bh = (size_t)b * CH + h;

    const unsigned short* qbase = qm + (bh * CS + (size_t)win * CWIN + q0w + ln) * CHD;
    bf16x16 qa0 = ld_a(qbase, half);
    bf16x16 qa1 = ld_a(qbase + 32, half);

    const unsigned short* kbase = km + (bh * CS + (size_t)win * CWIN) * CHD;
    const unsigned short* vbase = vt + bh * CHD * CS;   // row=hd, ld=CS

    f32x8 o0 = {}, o1 = {}, o2 = {}, o3 = {};
    float mrow[8], lrow[8];
#pragma unroll
    for (int r = 0; r < 8; ++r) { mrow[r] = -__builtin_inff(); lrow[r] = 0.f; }

    // ---- local causal window ----
    for (int kb = 0; kb < q0w + 16; kb += 32) {
        const unsigned short* kr0 = kbase + (size_t)(kb + ln) * CHD;
        const unsigned short* kr1 = kbase + (size_t)(kb + 16 + ln) * CHD;
        f32x8 s0 = {}, s1 = {};
        s0 = wmma_bf16(qa0, ld_b(kr0, half), s0);
        s0 = wmma_bf16(qa1, ld_b(kr0 + 32, half), s0);
        s1 = wmma_bf16(qa0, ld_b(kr1, half), s1);
        s1 = wmma_bf16(qa1, ld_b(kr1 + 32, half), s1);
        flash_block(s0, s1, true, kb, q0w, half, ln,
                    vbase, CS, win * CWIN + kb, mrow, lrow, o0, o1, o2, o3, lp[w]);
    }
#pragma unroll
    for (int r = 0; r < 8; ++r) {
        float inv = 1.0f / lrow[r];
        o0[r] *= inv; o1[r] *= inv; o2[r] *= inv; o3[r] *= inv;
        mrow[r] = -__builtin_inff(); lrow[r] = 0.f;
    }

    // ---- global tokens (separate softmax) ----
    f32x8 g0 = {}, g1 = {}, g2 = {}, g3 = {};
    const unsigned short* gkb = gk + bh * CG * CHD;
    const unsigned short* gvb = gvt + bh * CHD * CG;    // row=hd, ld=CG
    for (int kb = 0; kb < CG; kb += 32) {
        const unsigned short* kr0 = gkb + (size_t)(kb + ln) * CHD;
        const unsigned short* kr1 = gkb + (size_t)(kb + 16 + ln) * CHD;
        f32x8 s0 = {}, s1 = {};
        s0 = wmma_bf16(qa0, ld_b(kr0, half), s0);
        s0 = wmma_bf16(qa1, ld_b(kr0 + 32, half), s0);
        s1 = wmma_bf16(qa0, ld_b(kr1, half), s1);
        s1 = wmma_bf16(qa1, ld_b(kr1 + 32, half), s1);
        flash_block(s0, s1, false, kb, 0, half, ln,
                    gvb, CG, kb, mrow, lrow, g0, g1, g2, g3, lp[w]);
    }

    // ---- finalize: local + global, write bf16 [b, s, h*64+hd] ----
    size_t qg = (size_t)win * CWIN + q0w;
    unsigned short* ob = attn + ((size_t)b * CS + qg) * (CH * CHD) + h * CHD;
#pragma unroll
    for (int r = 0; r < 8; ++r) {
        float inv = 1.0f / lrow[r];
        unsigned short* orow = ob + (size_t)(half * 8 + r) * (CH * CHD);
        orow[0 + ln]  = f2bf(o0[r] + g0[r] * inv);
        orow[16 + ln] = f2bf(o1[r] + g1[r] * inv);
        orow[32 + ln] = f2bf(o2[r] + g2[r] * inv);
        orow[48 + ln] = f2bf(o3[r] + g3[r] * inv);
    }
}

// ---------------------------------------------------------------------------

extern "C" void kernel_launch(void* const* d_in, const int* in_sizes, int n_in,
                              void* d_out, int out_size, void* d_ws, size_t ws_size,
                              hipStream_t stream) {
    const float* x   = (const float*)d_in[0];
    const float* wq  = (const float*)d_in[1];
    const float* wk  = (const float*)d_in[2];
    const float* wv  = (const float*)d_in[3];
    const float* wo  = (const float*)d_in[4];
    const float* ws1 = (const float*)d_in[5];
    const float* bs1 = (const float*)d_in[6];
    const float* ws2 = (const float*)d_in[7];
    const float* bs2 = (const float*)d_in[8];
    float* out = (float*)d_out;

    char* ws = (char*)d_ws;
    size_t off = 0;
    auto alloc = [&](size_t bytes) -> void* {
        void* p = ws + off;
        off += (bytes + 255) & ~(size_t)255;
        return p;
    };
    const size_t NX = (size_t)CB * CS * CD;          // 8388608
    unsigned short* xbf   = (unsigned short*)alloc(NX * 2);
    unsigned short* wqt   = (unsigned short*)alloc((size_t)CD * 1024 * 2);
    unsigned short* wkt   = (unsigned short*)alloc((size_t)CD * 1024 * 2);
    unsigned short* wvt   = (unsigned short*)alloc((size_t)CD * 1024 * 2);
    unsigned short* wot   = (unsigned short*)alloc((size_t)1024 * CD * 2);
    unsigned short* qbf   = (unsigned short*)alloc(NX * 2);
    unsigned short* kbf   = (unsigned short*)alloc(NX * 2);
    unsigned short* vtb   = (unsigned short*)alloc(NX * 2);
    unsigned short* gkb   = (unsigned short*)alloc((size_t)CB * CH * CG * CHD * 2);
    unsigned short* gvtb  = (unsigned short*)alloc((size_t)CB * CH * CG * CHD * 2);
    unsigned short* gtok  = (unsigned short*)alloc((size_t)CB * CG * CD * 2);
    unsigned short* attnb = (unsigned short*)alloc(NX * 2);
    float* scores = (float*)alloc((size_t)CB * CS * 4);
    int*   idx    = (int*)alloc((size_t)CB * CG * 4);

    // 1. converts
    cvt_bf16_kernel<<<(int)(NX / 256), 256, 0, stream>>>(x, xbf, (int)NX);
    cvt_t_kernel<<<4096, 256, 0, stream>>>(wq, wqt, CD, 1024);
    cvt_t_kernel<<<4096, 256, 0, stream>>>(wk, wkt, CD, 1024);
    cvt_t_kernel<<<4096, 256, 0, stream>>>(wv, wvt, CD, 1024);
    cvt_t_kernel<<<4096, 256, 0, stream>>>(wo, wot, 1024, CD);

    // 2. token scores, 3. top-k, 4. gather
    score_kernel<<<CB * CS, 256, 0, stream>>>(x, ws1, bs1, ws2, bs2, scores);
    topk_kernel<<<(CB * CS) / 256, 256, 0, stream>>>(scores, idx);
    gather_kernel<<<(CB * CG * CD) / 256, 256, 0, stream>>>(xbf, idx, gtok);

    // 5-9. projections (WMMA GEMMs)
    auto gemm = [&](const unsigned short* A, const unsigned short* Bt, int M, int N,
                    int Kd, int mode, float scale, int S_, unsigned short* ob, float* of) {
        int blocks = (M / 16) * (N / 64) / 8;
        gemm_bf16_kernel<<<blocks, 256, 0, stream>>>(A, Bt, M, N, Kd, mode, scale, S_, ob, of);
    };
    const float SC = 0.125f;   // 1/sqrt(64)
    gemm(xbf,  wqt, CB * CS, 1024, CD, 0, SC,   CS, qbf,  nullptr);   // q (scaled)
    gemm(xbf,  wkt, CB * CS, 1024, CD, 0, 1.0f, CS, kbf,  nullptr);   // k
    gemm(xbf,  wvt, CB * CS, 1024, CD, 1, 1.0f, CS, vtb,  nullptr);   // v^T
    gemm(gtok, wkt, CB * CG, 1024, CD, 0, 1.0f, CG, gkb,  nullptr);   // gk
    gemm(gtok, wvt, CB * CG, 1024, CD, 1, 1.0f, CG, gvtb, nullptr);   // gv^T

    // 10. attention (local windows + global), 8192 query-tiles / 8 waves
    attn_kernel<<<1024, 256, 0, stream>>>(qbf, kbf, vtb, gkb, gvtb, attnb);

    // 11. output projection -> f32 d_out
    gemm(attnb, wot, CB * CS, CD, 1024, 2, 1.0f, 0, nullptr, out);
    (void)in_sizes; (void)n_in; (void)out_size; (void)ws_size;
}
